// MultiHeadAttention_31834297598688
// MI455X (gfx1250) — matrix operand order repaired and checked by reading, not compiled, not run on hardware
//
#include <hip/hip_runtime.h>
#include <hip/hip_bf16.h>
#include <stdint.h>

// Problem constants (match reference)
#define B_   4
#define LQ_  1024
#define LK_  2048
#define DM_  1024    // model dim (D_IN = D_KEY = D_VAL = D_OUT)
#define H_   16
#define DH_  64      // head dim

typedef __attribute__((ext_vector_type(16))) __bf16        bf16x16;
typedef __attribute__((ext_vector_type(8)))  float         f32x8;
typedef __attribute__((ext_vector_type(8)))  unsigned int  u32x8;
typedef __attribute__((ext_vector_type(4)))  unsigned int  u32x4;

static __device__ __forceinline__ f32x8 zero8() {
    f32x8 z = {0.f,0.f,0.f,0.f,0.f,0.f,0.f,0.f};
    return z;
}

static __device__ __forceinline__ f32x8 wmma_bf16(bf16x16 a, bf16x16 b, f32x8 c) {
    // D = A(16x32 bf16) * B(32x16 bf16) + C(16x16 f32)
    return __builtin_amdgcn_wmma_f32_16x16x32_bf16(
        /*neg_a=*/false, a, /*neg_b=*/false, b,
        /*c_mod=*/(short)0, c, /*reuse_a=*/false, /*reuse_b=*/false);
}

// ---------------------------------------------------------------------------
// CDNA5 async global->LDS staging (ASYNCcnt-tracked, no VGPR data movement).
// LDS byte offset obtained via addrspace(3) ptrtoint.
// ---------------------------------------------------------------------------
typedef __attribute__((address_space(3))) void lds_void_t;

static __device__ __forceinline__ unsigned lds_off(const void* p) {
    return (unsigned)(unsigned long long)(lds_void_t*)const_cast<void*>(p);
}

static __device__ __forceinline__ void async_copy_b128(const void* lds_dst, const void* gsrc) {
    unsigned l = lds_off(lds_dst);
    asm volatile("global_load_async_to_lds_b128 %0, %1, off"
                 :: "v"(l), "v"(gsrc) : "memory");
}

static __device__ __forceinline__ void async_copy_b64(const void* lds_dst, const void* gsrc) {
    unsigned l = lds_off(lds_dst);
    asm volatile("global_load_async_to_lds_b64 %0, %1, off"
                 :: "v"(l), "v"(gsrc) : "memory");
}

static __device__ __forceinline__ void wait_async0() {
    asm volatile("s_wait_asynccnt 0x0" ::: "memory");
}

// A fragment (16 rows x 32 K, bf16) from row-major buffer, row stride `ld` elements.
// ISA layout: lane L holds row m=L%16; VGPR0-3 = K 8g..8g+7, VGPR4-7 = K 16+8g..16+8g+7 (g=L/16).
static __device__ __forceinline__ bf16x16 load_frag_a(const __bf16* p, int ld) {
    const int lane = threadIdx.x & 31;
    const __bf16* r = p + (lane & 15) * ld + ((lane >> 4) * 8);
    u32x4 lo = *(const u32x4*)(r);        // K = 8g .. 8g+7   (16B)
    u32x4 hi = *(const u32x4*)(r + 16);   // K = 16+8g .. +7  (16B)
    u32x8 v;
    v[0]=lo[0]; v[1]=lo[1]; v[2]=lo[2]; v[3]=lo[3];
    v[4]=hi[0]; v[5]=hi[1]; v[6]=hi[2]; v[7]=hi[3];
    return __builtin_bit_cast(bf16x16, v);
}

// B fragment (32 K x 16 N, bf16) read from a *transposed* buffer T[n][k], row stride `ld`.
// ISA layout: lane L holds col n=L%16; VGPR0-3 = K 8g..8g+7, VGPR4-7 = K 16+8g..16+8g+7 (g=L/16).
static __device__ __forceinline__ bf16x16 load_frag_bT(const __bf16* p, int ld) {
    const int lane = threadIdx.x & 31;
    const __bf16* r = p + (lane & 15) * ld + ((lane >> 4) * 8);
    u32x4 lo = *(const u32x4*)(r);        // k = 8g .. 8g+7
    u32x4 hi = *(const u32x4*)(r + 16);   // k = 16+8g .. +7
    u32x8 v;
    v[0]=lo[0]; v[1]=lo[1]; v[2]=lo[2]; v[3]=lo[3];
    v[4]=hi[0]; v[5]=hi[1]; v[6]=hi[2]; v[7]=hi[3];
    return __builtin_bit_cast(bf16x16, v);
}

// ---------------------------------------------------------------------------
// f32 -> bf16 conversion (grid-stride)
// ---------------------------------------------------------------------------
__global__ __launch_bounds__(256)
void cvt_f32_bf16(const float* __restrict__ x, __bf16* __restrict__ y, size_t n) {
    size_t i = (size_t)blockIdx.x * blockDim.x + threadIdx.x;
    size_t stride = (size_t)gridDim.x * blockDim.x;
    for (; i < n; i += stride) y[i] = (__bf16)x[i];
}

// ---------------------------------------------------------------------------
// Tiled bf16 GEMM: C = A[M,1024] * W[1024,1024] + bias
// MODE 0: write bf16 in head-split layout out[((b*H+h)*Lrows + r)*64 + d]
// MODE 1: write f32 row-major
// Block 256 threads (8 waves), tile 64x64, K-step 32.
// A tile staged with async-to-LDS; W tile staged manually (needs transpose).
// ---------------------------------------------------------------------------
template <int MODE>
__global__ __launch_bounds__(256)
void gemm_bf16_k(const __bf16* __restrict__ A, const __bf16* __restrict__ W,
                 const float* __restrict__ bias, __bf16* __restrict__ outh,
                 float* __restrict__ outf, int M, int Lrows)
{
    __shared__ __attribute__((aligned(16))) __bf16 As[64 * 32];   // As[m][k]
    __shared__ __attribute__((aligned(16))) __bf16 Bt[64 * 32];   // Bt[n][k] (transposed)

    const int m0 = blockIdx.x * 64;
    const int n0 = blockIdx.y * 64;
    const int t  = threadIdx.x;
    const int w  = t >> 5;
    const int lane = t & 31;
    const int mSub = w & 3;
    const int nSubBase = (w >> 2) * 2;   // each wave: 2 N-subtiles

    f32x8 acc0 = zero8(), acc1 = zero8();

    for (int k0 = 0; k0 < DM_; k0 += 32) {
        __syncthreads();
        {   // stage A tile 64x32: per-lane 16B contiguous -> async-to-LDS b128
            int row = t >> 2, c8 = (t & 3) * 8;
            async_copy_b128(&As[row * 32 + c8],
                            &A[(size_t)(m0 + row) * DM_ + k0 + c8]);
        }
        {   // stage B tile 32x64 transposed: Bt[n][k] = W[k0+k][n0+n]
            int kk = t >> 3, n8 = (t & 7) * 8;
            u32x4 raw = *(const u32x4*)&W[(size_t)(k0 + kk) * DM_ + n0 + n8];
            const __bf16* e = (const __bf16*)&raw;
            #pragma unroll
            for (int i = 0; i < 8; ++i) Bt[(n8 + i) * 32 + kk] = e[i];
        }
        wait_async0();
        __syncthreads();

        bf16x16 a  = load_frag_a(&As[mSub * 16 * 32], 32);
        bf16x16 b0 = load_frag_bT(&Bt[(nSubBase * 16) * 32], 32);
        bf16x16 b1 = load_frag_bT(&Bt[((nSubBase + 1) * 16) * 32], 32);
        acc0 = wmma_bf16(a, b0, acc0);
        acc1 = wmma_bf16(a, b1, acc1);
    }

    const int nl = lane & 15, half = lane >> 4;
    #pragma unroll
    for (int s = 0; s < 2; ++s) {
        f32x8 acc = s ? acc1 : acc0;
        const int n_g = n0 + (nSubBase + s) * 16 + nl;
        const float bvv = bias[n_g];
        #pragma unroll
        for (int j = 0; j < 8; ++j) {
            const int m_g = m0 + mSub * 16 + j + 8 * half;
            const float v = acc[j] + bvv;
            if (MODE == 0) {
                const int bI = m_g / Lrows, r = m_g % Lrows;
                const int h = n_g >> 6, d = n_g & 63;
                outh[(((size_t)bI * H_ + h) * Lrows + r) * DH_ + d] = (__bf16)v;
            } else {
                outf[(size_t)m_g * DM_ + n_g] = v;
            }
        }
    }
}

// ---------------------------------------------------------------------------
// Attention pass 1: per-row softmax stats (m, l) with online max/sum.
// Grid (B*H, LQ/128), block 256 (8 waves x 16 q-rows).
// ---------------------------------------------------------------------------
__global__ __launch_bounds__(256)
void attn_stats_k(const __bf16* __restrict__ Qh, const __bf16* __restrict__ Kh,
                  const int* __restrict__ mask,
                  float* __restrict__ Mb, float* __restrict__ Lb)
{
    __shared__ __attribute__((aligned(16))) __bf16 Ks[16 * 64];   // Ks[key][dk]

    const int bh = blockIdx.x, b = bh >> 4;
    const int t = threadIdx.x, w = t >> 5, lane = t & 31;
    const int qw = blockIdx.y * 128 + w * 16;
    const int nl = lane & 15, half = lane >> 4;

    const __bf16* Qbase = Qh + ((size_t)bh * LQ_ + qw) * DH_;
    const __bf16* Kbase = Kh + (size_t)bh * LK_ * DH_;

    bf16x16 qa0 = load_frag_a(Qbase, DH_);
    bf16x16 qa1 = load_frag_a(Qbase + 32, DH_);

    float mrun[8], lrun[8];
    #pragma unroll
    for (int j = 0; j < 8; ++j) { mrun[j] = -3.0e38f; lrun[j] = 0.f; }

    for (int k0 = 0; k0 < LK_; k0 += 16) {
        __syncthreads();
        {   // stage 16x64 K tile: 256 lanes x 8B -> async-to-LDS b64
            int row = t >> 4, c4 = (t & 15) * 4;
            async_copy_b64(&Ks[row * 64 + c4],
                           &Kbase[(size_t)(k0 + row) * DH_ + c4]);
        }
        wait_async0();
        __syncthreads();

        f32x8 s = zero8();
        s = wmma_bf16(qa0, load_frag_bT(Ks, 64), s);          // dk 0..31
        s = wmma_bf16(qa1, load_frag_bT(Ks + 32, 64), s);     // dk 32..63

        const int kg = k0 + nl;
        #pragma unroll
        for (int j = 0; j < 8; ++j) {
            const int qg = qw + j + 8 * half;
            const int mv = mask[((size_t)b * LQ_ + qg) * LK_ + kg];
            float x = (mv == 0) ? -1e9f : s[j] * 0.125f;      // / sqrt(64)
            float tmax = x;
            tmax = fmaxf(tmax, __shfl_xor(tmax, 1));
            tmax = fmaxf(tmax, __shfl_xor(tmax, 2));
            tmax = fmaxf(tmax, __shfl_xor(tmax, 4));
            tmax = fmaxf(tmax, __shfl_xor(tmax, 8));
            const float newm = fmaxf(mrun[j], tmax);
            float e = __expf(x - newm);
            e += __shfl_xor(e, 1); e += __shfl_xor(e, 2);
            e += __shfl_xor(e, 4); e += __shfl_xor(e, 8);
            lrun[j] = lrun[j] * __expf(mrun[j] - newm) + e;
            mrun[j] = newm;
        }
    }

    if (nl == 0) {
        #pragma unroll
        for (int j = 0; j < 8; ++j) {
            const int qg = qw + j + 8 * half;
            const size_t idx = (size_t)bh * LQ_ + qg;
            Mb[idx] = mrun[j];
            Lb[idx] = lrun[j];
        }
    }
}

// ---------------------------------------------------------------------------
// Attention pass 2: O = softmax(S) * V per head (stats precomputed).
// Grid (B*H, LQ/128), block 256 (8 waves x 16 q-rows). Key step 32.
// K tile async-to-LDS; V tile manual (needs transpose).
// ---------------------------------------------------------------------------
__global__ __launch_bounds__(256)
void attn_out_k(const __bf16* __restrict__ Qh, const __bf16* __restrict__ Kh,
                const __bf16* __restrict__ Vh, const int* __restrict__ mask,
                const float* __restrict__ Mb, const float* __restrict__ Lb,
                __bf16* __restrict__ WV)
{
    __shared__ __attribute__((aligned(16))) __bf16 Ks[32 * 64];      // Ks[key][dk]
    __shared__ __attribute__((aligned(16))) __bf16 Vt[64 * 32];      // Vt[dv][key]
    __shared__ __attribute__((aligned(16))) __bf16 Ps[8][16 * 32];   // per-wave P tile

    const int bh = blockIdx.x, b = bh >> 4, h = bh & 15;
    const int t = threadIdx.x, w = t >> 5, lane = t & 31;
    const int qw = blockIdx.y * 128 + w * 16;
    const int nl = lane & 15, half = lane >> 4;

    const __bf16* Qbase = Qh + ((size_t)bh * LQ_ + qw) * DH_;
    const __bf16* Kbase = Kh + (size_t)bh * LK_ * DH_;
    const __bf16* Vbase = Vh + (size_t)bh * LK_ * DH_;

    bf16x16 qa0 = load_frag_a(Qbase, DH_);
    bf16x16 qa1 = load_frag_a(Qbase + 32, DH_);

    float mj[8], li[8];
    #pragma unroll
    for (int j = 0; j < 8; ++j) {
        const int qg = qw + j + 8 * half;
        const size_t idx = (size_t)bh * LQ_ + qg;
        mj[j] = Mb[idx];
        li[j] = 1.f / Lb[idx];
    }

    f32x8 o0 = zero8(), o1 = zero8(), o2 = zero8(), o3 = zero8();

    for (int k0 = 0; k0 < LK_; k0 += 32) {
        __syncthreads();
        {   // stage 32x64 K tile: 256 lanes x 16B -> async-to-LDS b128
            int row = t >> 3, c8 = (t & 7) * 8;
            async_copy_b128(&Ks[row * 64 + c8],
                            &Kbase[(size_t)(k0 + row) * DH_ + c8]);
        }
        {   // stage V transposed: Vt[dv][key]
            int key = t >> 3, d8 = (t & 7) * 8;
            u32x4 raw = *(const u32x4*)&Vbase[(size_t)(k0 + key) * DH_ + d8];
            const __bf16* e = (const __bf16*)&raw;
            #pragma unroll
            for (int i = 0; i < 8; ++i) Vt[(d8 + i) * 32 + key] = e[i];
        }
        wait_async0();
        __syncthreads();

        // two 16-key score subtiles -> masked softmax weights -> Ps (bf16, A layout source)
        #pragma unroll
        for (int kk = 0; kk < 32; kk += 16) {
            f32x8 s = zero8();
            s = wmma_bf16(qa0, load_frag_bT(&Ks[kk * 64], 64), s);
            s = wmma_bf16(qa1, load_frag_bT(&Ks[kk * 64 + 32], 64), s);
            const int kg = k0 + kk + nl;
            #pragma unroll
            for (int j = 0; j < 8; ++j) {
                const int qg = qw + j + 8 * half;
                const int mv = mask[((size_t)b * LQ_ + qg) * LK_ + kg];
                const float wt = (mv == 0) ? 0.f
                               : __expf(s[j] * 0.125f - mj[j]) * li[j];
                Ps[w][(j + 8 * half) * 32 + kk + nl] = (__bf16)wt;
            }
        }

        // P(16x32) * V(32x64): 4 dv-subtiles
        bf16x16 pa = load_frag_a(&Ps[w][0], 32);
        o0 = wmma_bf16(pa, load_frag_bT(&Vt[0 * 16 * 32], 32), o0);
        o1 = wmma_bf16(pa, load_frag_bT(&Vt[1 * 16 * 32], 32), o1);
        o2 = wmma_bf16(pa, load_frag_bT(&Vt[2 * 16 * 32], 32), o2);
        o3 = wmma_bf16(pa, load_frag_bT(&Vt[3 * 16 * 32], 32), o3);
    }

    #pragma unroll
    for (int d = 0; d < 4; ++d) {
        f32x8 o = (d == 0) ? o0 : (d == 1) ? o1 : (d == 2) ? o2 : o3;
        #pragma unroll
        for (int j = 0; j < 8; ++j) {
            const int qg = qw + j + 8 * half;
            WV[((size_t)b * LQ_ + qg) * DM_ + h * DH_ + d * 16 + nl] = (__bf16)o[j];
        }
    }
}

// ---------------------------------------------------------------------------
// Attention pass 3: weight_mean = (1/H) * sum_h softmax weights.
// Head loop INSIDE the workgroup -> register accumulation, no atomics.
// Grid (B, LQ/64, LK/256), block 128 (4 waves x 16 q-rows); 16 k-tiles of 16.
// ---------------------------------------------------------------------------
__global__ __launch_bounds__(128)
void attn_weight_k(const __bf16* __restrict__ Qh, const __bf16* __restrict__ Kh,
                   const int* __restrict__ mask,
                   const float* __restrict__ Mb, const float* __restrict__ Lb,
                   float* __restrict__ Wout)
{
    __shared__ __attribute__((aligned(16))) __bf16 Ks[16 * 64];

    const int b = blockIdx.x;
    const int t = threadIdx.x, w = t >> 5, lane = t & 31;
    const int qw = blockIdx.y * 64 + w * 16;
    const int k0 = blockIdx.z * 256;
    const int nl = lane & 15, half = lane >> 4;

    f32x8 acc[16];
    #pragma unroll
    for (int i = 0; i < 16; ++i) acc[i] = zero8();

    for (int h = 0; h < H_; ++h) {
        const int bh = b * H_ + h;
        const __bf16* Qbase = Qh + ((size_t)bh * LQ_ + qw) * DH_;
        const __bf16* Kbase = Kh + (size_t)bh * LK_ * DH_;

        bf16x16 qa0 = load_frag_a(Qbase, DH_);
        bf16x16 qa1 = load_frag_a(Qbase + 32, DH_);

        float mj[8], li[8];
        #pragma unroll
        for (int j = 0; j < 8; ++j) {
            const int qg = qw + j + 8 * half;
            const size_t idx = (size_t)bh * LQ_ + qg;
            mj[j] = Mb[idx];
            li[j] = 1.f / Lb[idx];
        }

        #pragma unroll
        for (int kt = 0; kt < 16; ++kt) {
            __syncthreads();
            {   // stage 16x64 K tile: 128 lanes x 16B -> async-to-LDS b128
                int row = t >> 3, c8 = (t & 7) * 8;
                async_copy_b128(&Ks[row * 64 + c8],
                                &Kbase[(size_t)(k0 + kt * 16 + row) * DH_ + c8]);
            }
            wait_async0();
            __syncthreads();

            f32x8 s = zero8();
            s = wmma_bf16(qa0, load_frag_bT(Ks, 64), s);
            s = wmma_bf16(qa1, load_frag_bT(Ks + 32, 64), s);

            const int kg = k0 + kt * 16 + nl;
            #pragma unroll
            for (int j = 0; j < 8; ++j) {
                const int qg = qw + j + 8 * half;
                const int mv = mask[((size_t)b * LQ_ + qg) * LK_ + kg];
                const float wt = (mv == 0) ? 0.f
                               : __expf(s[j] * 0.125f - mj[j]) * li[j];
                acc[kt][j] += wt;
            }
        }
    }

    #pragma unroll
    for (int kt = 0; kt < 16; ++kt) {
        #pragma unroll
        for (int j = 0; j < 8; ++j) {
            const int qg = qw + j + 8 * half;
            Wout[((size_t)b * LQ_ + qg) * LK_ + k0 + kt * 16 + nl] = acc[kt][j] * 0.0625f;
        }
    }
}

// ---------------------------------------------------------------------------
// Host launcher
// ---------------------------------------------------------------------------
extern "C" void kernel_launch(void* const* d_in, const int* in_sizes, int n_in,
                              void* d_out, int out_size, void* d_ws, size_t ws_size,
                              hipStream_t stream)
{
    (void)in_sizes; (void)n_in; (void)out_size; (void)ws_size;

    const float* X    = (const float*)d_in[0];   // [B,LQ,1024]
    const float* R    = (const float*)d_in[1];   // [B,LK,1024]
    const int*   mask = (const int*)  d_in[2];   // [B,LQ,LK]
    const float* Wq   = (const float*)d_in[3];
    const float* bq   = (const float*)d_in[4];
    const float* Wk   = (const float*)d_in[5];
    const float* bk   = (const float*)d_in[6];
    const float* Wv   = (const float*)d_in[7];
    const float* bv   = (const float*)d_in[8];
    const float* Wo   = (const float*)d_in[9];
    const float* bo   = (const float*)d_in[10];

    float* out  = (float*)d_out;                        // [B,LQ,1024]
    float* wout = out + (size_t)B_ * LQ_ * DM_;         // [B,LQ,LK]

    // Workspace carve-out (256B aligned)
    char* ws = (char*)d_ws;
    auto carve = [&](size_t bytes) {
        char* p = ws;
        ws += (bytes + 255) & ~(size_t)255;
        return p;
    };
    const size_t nX  = (size_t)B_ * LQ_ * DM_;
    const size_t nR  = (size_t)B_ * LK_ * DM_;
    const size_t nW  = (size_t)DM_ * DM_;
    __bf16* Xb  = (__bf16*)carve(nX * 2);
    __bf16* Rb  = (__bf16*)carve(nR * 2);
    __bf16* Wqb = (__bf16*)carve(nW * 2);
    __bf16* Wkb = (__bf16*)carve(nW * 2);
    __bf16* Wvb = (__bf16*)carve(nW * 2);
    __bf16* Wob = (__bf16*)carve(nW * 2);
    __bf16* Qh  = (__bf16*)carve((size_t)B_ * H_ * LQ_ * DH_ * 2);
    __bf16* Kh  = (__bf16*)carve((size_t)B_ * H_ * LK_ * DH_ * 2);
    __bf16* Vh  = (__bf16*)carve((size_t)B_ * H_ * LK_ * DH_ * 2);
    __bf16* WV  = (__bf16*)carve(nX * 2);
    float*  Mb  = (float*)carve((size_t)B_ * H_ * LQ_ * 4);
    float*  Lb  = (float*)carve((size_t)B_ * H_ * LQ_ * 4);

    // 1. f32 -> bf16 conversions
    cvt_f32_bf16<<<4096, 256, 0, stream>>>(X,  Xb,  nX);
    cvt_f32_bf16<<<4096, 256, 0, stream>>>(R,  Rb,  nR);
    cvt_f32_bf16<<<2048, 256, 0, stream>>>(Wq, Wqb, nW);
    cvt_f32_bf16<<<2048, 256, 0, stream>>>(Wk, Wkb, nW);
    cvt_f32_bf16<<<2048, 256, 0, stream>>>(Wv, Wvb, nW);
    cvt_f32_bf16<<<2048, 256, 0, stream>>>(Wo, Wob, nW);

    // 2. Projections -> head-split bf16 [B,H,L,64]
    dim3 gq(B_ * LQ_ / 64, DM_ / 64);     // (64,16)
    dim3 gr(B_ * LK_ / 64, DM_ / 64);     // (128,16)
    gemm_bf16_k<0><<<gq, 256, 0, stream>>>(Xb, Wqb, bq, Qh, nullptr, B_ * LQ_, LQ_);
    gemm_bf16_k<0><<<gr, 256, 0, stream>>>(Rb, Wkb, bk, Kh, nullptr, B_ * LK_, LK_);
    gemm_bf16_k<0><<<gr, 256, 0, stream>>>(Rb, Wvb, bv, Vh, nullptr, B_ * LK_, LK_);

    // 3. Softmax stats (m, l)
    attn_stats_k<<<dim3(B_ * H_, LQ_ / 128), 256, 0, stream>>>(Qh, Kh, mask, Mb, Lb);

    // 4. O = softmax(S) V  -> merged heads WV [B*LQ, 1024] bf16
    attn_out_k<<<dim3(B_ * H_, LQ_ / 128), 256, 0, stream>>>(Qh, Kh, Vh, mask, Mb, Lb, WV);

    // 5. weight_mean (register-accumulated over heads, no atomics)
    attn_weight_k<<<dim3(B_, LQ_ / 64, LK_ / 256), 128, 0, stream>>>(Qh, Kh, mask, Mb, Lb, wout);

    // 6. Final projection: out = WV @ Wo + bo (f32)
    gemm_bf16_k<1><<<gq, 256, 0, stream>>>(WV, Wob, bo, nullptr, out, B_ * LQ_, LQ_);
}
